// Mask2FormerCriterion_24859270709859
// MI455X (gfx1250) — matrix operand order repaired and checked by reading, not compiled, not run on hardware
//
#include <hip/hip_runtime.h>

typedef __attribute__((ext_vector_type(16))) _Float16 v16h;
typedef __attribute__((ext_vector_type(8)))  float    v8f;
typedef __attribute__((ext_vector_type(4)))  float    v4f;

#define NUM_CLASSES 19
#define NB 8
#define NQ 100
#define QP 112            // Q padded to 7*16
#define CP 32             // classes padded to 2 WMMA column tiles
#define HWX 32768         // 128*256
#define NL 3

__device__ __forceinline__ float fexp2(float x) { return __builtin_amdgcn_exp2f(x); }
__device__ __forceinline__ float flog2(float x) { return __builtin_amdgcn_logf(x); }
__device__ __forceinline__ float frcp(float x)  { return __builtin_amdgcn_rcpf(x); }
#define LOG2E 1.4426950408889634f
#define LN2   0.6931471805599453f

// ---------------------------------------------------------------------------
// Kernel 1: per-batch class histogram of anno -> gsum[b][32] (float)
// ---------------------------------------------------------------------------
__global__ __launch_bounds__(256) void gsum_kernel(const int* __restrict__ anno,
                                                   float* __restrict__ gsum) {
  int b = blockIdx.x;
  __shared__ int hist[CP];
  if (threadIdx.x < CP) hist[threadIdx.x] = 0;
  __syncthreads();
  const int* a = anno + (size_t)b * HWX;
  for (int i = threadIdx.x; i < HWX; i += 256) {
    int c = a[i];
    if (c >= 0 && c < NUM_CLASSES) atomicAdd(&hist[c], 1);
  }
  __syncthreads();
  if (threadIdx.x < CP) gsum[b * CP + threadIdx.x] = (float)hist[threadIdx.x];
}

// ---------------------------------------------------------------------------
// Kernel 2: single pass over all mask tensors (the bandwidth-bound part).
// For each (layer l, batch b, 16-row query tile rt) computes via WMMA:
//   dot_pg[l,b,q,c]  = sum_hw p[q,hw]      * (anno[hw]==c)
//   dot_sg[l,b,q,c]  = sum_hw sigmoid(p)   * (anno[hw]==c)
//   sp_sum[l,b,q]    = sum_hw softplus(p)
//   sig_sum[l,b,q]   = sum_hw sigmoid(p)
// 8 waves per block split the K=HW range; LDS float atomics reduce.
// Padding rows (q>=100) load a clamped row unguarded; their WMMA output rows
// are simply never stored (no EXEC divergence in the hot loop).
// ---------------------------------------------------------------------------
__global__ __launch_bounds__(256) void stats_kernel(
    const float* __restrict__ m0, const float* __restrict__ m1,
    const float* __restrict__ m2, const int* __restrict__ anno,
    float* __restrict__ dot_pg, float* __restrict__ dot_sg,
    float* __restrict__ sp_sum, float* __restrict__ sig_sum) {
  const int rt = blockIdx.x;   // 0..6  query row tile
  const int b  = blockIdx.y;   // 0..7
  const int l  = blockIdx.z;   // 0..2
  const float* masks = (l == 0) ? m0 : (l == 1) ? m1 : m2;

  const int tid  = threadIdx.x;
  const int wave = tid >> 5;
  const int lane = tid & 31;
  const int row  = lane & 15;        // A-matrix row within tile
  const int half = lane >> 4;        // K-half select per CDNA5 16-bit A layout
  const int q0   = rt * 16;
  const int q    = q0 + row;
  const bool qvalid = (q < NQ);
  const int qc   = qvalid ? q : (NQ - 1);   // clamped row: always-legal loads

  const float* prow = masks + (size_t)(b * NQ + qc) * HWX;
  const int*   arow = anno + (size_t)b * HWX;

  v8f accPG0 = {}, accPG1 = {}, accSG0 = {}, accSG1 = {};
  float sp_part = 0.f, sig_part = 0.f;

  const int kbeg = wave * (HWX / 8);
  const int kend = kbeg + (HWX / 8);
  for (int k0 = kbeg; k0 < kend; k0 += 32) {
    // A-tile K indices for this lane are two contiguous 8-float runs:
    //   [k0+8h, k0+8h+7]  (VGPR pairs v=0..3)  and
    //   [k0+16+8h, ...+7] (VGPR pairs v=4..7)
    const v4f* s0 = (const v4f*)(prow + k0 + 8 * half);
    const v4f* s1 = (const v4f*)(prow + k0 + 16 + 8 * half);
    v4f c0 = __builtin_nontemporal_load(s0);
    v4f c1 = __builtin_nontemporal_load(s0 + 1);
    v4f c2 = __builtin_nontemporal_load(s1);
    v4f c3 = __builtin_nontemporal_load(s1 + 1);
    float xs[16] = {c0.x, c0.y, c0.z, c0.w, c1.x, c1.y, c1.z, c1.w,
                    c2.x, c2.y, c2.z, c2.w, c3.x, c3.y, c3.z, c3.w};

    v16h aP, aS;
#pragma unroll
    for (int j = 0; j < 16; ++j) {
      float x  = xs[j];
      float t  = fexp2(fabsf(x) * -LOG2E);        // exp(-|x|), one v_exp_f32
      float r  = frcp(1.0f + t);                  // sigmoid(|x|), v_rcp_f32
      float sg = (x >= 0.f) ? r : 1.0f - r;
      float sp = fmaxf(x, 0.f) + LN2 * flog2(1.0f + t);  // stable softplus
      sp_part  += sp;
      sig_part += sg;
      aP[j] = (_Float16)x;
      aS[j] = (_Float16)sg;
    }

    // B tile: 32x16 one-hot from anno. Lane%16 = class column; 16 vector elems
    // = K rows in the same order as A ([k0+8h..+7], [k0+16+8h..+7]).
    const int* ab0 = arow + k0 + 8 * half;
    const int* ab1 = arow + k0 + 16 + 8 * half;
    v16h b0t, b1t;
#pragma unroll
    for (int j = 0; j < 16; ++j) {
      int av = (j < 8) ? ab0[j] : ab1[j - 8];
      b0t[j] = (_Float16)((av == row) ? 1.0f : 0.0f);
      b1t[j] = (_Float16)((av == (16 + row)) ? 1.0f : 0.0f);
    }
    accPG0 = __builtin_amdgcn_wmma_f32_16x16x32_f16(false, aP, false, b0t,
                                                    (short)0, accPG0, false, false);
    accPG1 = __builtin_amdgcn_wmma_f32_16x16x32_f16(false, aP, false, b1t,
                                                    (short)0, accPG1, false, false);
    accSG0 = __builtin_amdgcn_wmma_f32_16x16x32_f16(false, aS, false, b0t,
                                                    (short)0, accSG0, false, false);
    accSG1 = __builtin_amdgcn_wmma_f32_16x16x32_f16(false, aS, false, b1t,
                                                    (short)0, accSG1, false, false);
  }

  // Cross-wave reduction in LDS (ds_add_f32 atomics)
  __shared__ float lds_pg[16][CP];
  __shared__ float lds_sg[16][CP];
  __shared__ float lds_sp[16];
  __shared__ float lds_sig[16];
  for (int i = tid; i < 16 * CP; i += 256) {
    ((float*)lds_pg)[i] = 0.f;
    ((float*)lds_sg)[i] = 0.f;
  }
  if (tid < 16) { lds_sp[tid] = 0.f; lds_sig[tid] = 0.f; }
  __syncthreads();

  const int n = lane & 15;  // C/D layout: N = lane%16, M = vgpr + 8*(lane/16)
#pragma unroll
  for (int i = 0; i < 8; ++i) {
    int m = i + 8 * half;
    atomicAdd(&lds_pg[m][n],      accPG0[i]);
    atomicAdd(&lds_pg[m][16 + n], accPG1[i]);
    atomicAdd(&lds_sg[m][n],      accSG0[i]);
    atomicAdd(&lds_sg[m][16 + n], accSG1[i]);
  }
  if (qvalid) {
    atomicAdd(&lds_sp[row],  sp_part);
    atomicAdd(&lds_sig[row], sig_part);
  }
  __syncthreads();

  const size_t base = (size_t)(l * NB + b) * QP;
  for (int i = tid; i < 16 * CP; i += 256) {
    int m = i >> 5, c = i & (CP - 1);
    if (q0 + m < NQ) {
      dot_pg[(base + q0 + m) * CP + c] = lds_pg[m][c];
      dot_sg[(base + q0 + m) * CP + c] = lds_sg[m][c];
    }
  }
  if (tid < 16 && q0 + tid < NQ) {
    sp_sum[base + q0 + tid]  = lds_sp[tid];
    sig_sum[base + q0 + tid] = lds_sig[tid];
  }
}

// ---------------------------------------------------------------------------
// Kernel 3: Hungarian matching per batch (from layer-2 stats).
// 100 threads build the [M<=19, 100] cost matrix, thread 0 solves (JV).
// ---------------------------------------------------------------------------
__global__ __launch_bounds__(128) void match_kernel(
    const float* __restrict__ cls2, const float* __restrict__ dot_pg,
    const float* __restrict__ dot_sg, const float* __restrict__ sp_sum,
    const float* __restrict__ sig_sum, const float* __restrict__ gsum,
    int* __restrict__ tgt, float* __restrict__ matched,
    float* __restrict__ msum) {
  const int b = blockIdx.x;
  __shared__ float cost[NUM_CLASSES][NQ];
  __shared__ int   present[NUM_CLASSES];
  __shared__ int   Msh;
  __shared__ float u[NUM_CLASSES + 1], v[NQ + 1], minv[NQ + 1];
  __shared__ int   p[NQ + 1], way[NQ + 1];
  __shared__ unsigned char used[NQ + 1];
  const int tid = threadIdx.x;

  if (tid == 0) {
    int M = 0;
    for (int c = 0; c < NUM_CLASSES; ++c)
      if (gsum[b * CP + c] > 0.5f) present[M++] = c;
    Msh = M;
  }
  __syncthreads();
  const int M = Msh;

  for (int q = tid; q < NQ; q += 128) {
    tgt[b * NQ + q] = NUM_CLASSES;
    matched[b * NQ + q] = 0.f;
  }
  if (tid == 0) msum[b] = (float)M;
  if (M == 0) return;

  const size_t sbase = (size_t)(2 * NB + b) * QP;  // layer-2 stats
  for (int q = tid; q < NQ; q += 128) {
    const float* lg = cls2 + (size_t)(b * NQ + q) * (NUM_CLASSES + 1);
    float mx = lg[0];
    for (int c = 1; c < NUM_CLASSES + 1; ++c) mx = fmaxf(mx, lg[c]);
    float se = 0.f;
    for (int c = 0; c < NUM_CLASSES + 1; ++c) se += fexp2((lg[c] - mx) * LOG2E);
    float inv = frcp(se);
    float spv = sp_sum[sbase + q];
    float sgv = sig_sum[sbase + q];
    for (int j = 0; j < M; ++j) {
      int c = present[j];
      float prob = fexp2((lg[c] - mx) * LOG2E) * inv;
      float bce  = (spv - dot_pg[(sbase + q) * CP + c]) * (1.0f / HWX);
      float dice = 1.f - 2.f * dot_sg[(sbase + q) * CP + c] /
                             (sgv + gsum[b * CP + c] + 1e-5f);
      cost[j][q] = 2.f * (-prob) + 5.f * bce + 5.f * dice;
    }
  }
  __syncthreads();

  if (tid == 0) {
    const float INF = 1e30f;
    for (int j = 0; j <= NQ; ++j) { v[j] = 0.f; p[j] = 0; }
    for (int i = 0; i <= M; ++i) u[i] = 0.f;
    for (int i = 1; i <= M; ++i) {
      p[0] = i;
      int j0 = 0;
      for (int j = 0; j <= NQ; ++j) { minv[j] = INF; used[j] = 0; }
      while (true) {
        used[j0] = 1;
        int i0 = p[j0];
        float delta = INF;
        int j1 = 0;
        for (int j = 1; j <= NQ; ++j) {
          if (!used[j]) {
            float cur = cost[i0 - 1][j - 1] - u[i0] - v[j];
            if (cur < minv[j]) { minv[j] = cur; way[j] = j0; }
            if (minv[j] < delta) { delta = minv[j]; j1 = j; }
          }
        }
        for (int j = 0; j <= NQ; ++j) {
          if (used[j]) { u[p[j]] += delta; v[j] -= delta; }
          else minv[j] -= delta;
        }
        j0 = j1;
        if (p[j0] == 0) break;
      }
      while (j0) { int jn = way[j0]; p[j0] = p[jn]; j0 = jn; }
    }
    for (int j = 1; j <= NQ; ++j) {
      int i = p[j];
      if (i > 0) {
        tgt[b * NQ + (j - 1)] = present[i - 1];
        matched[b * NQ + (j - 1)] = 1.f;
      }
    }
  }
}

// ---------------------------------------------------------------------------
// Kernel 4: fold everything into the final scalar loss (one block).
// ---------------------------------------------------------------------------
__global__ __launch_bounds__(256) void loss_kernel(
    const float* __restrict__ cls0, const float* __restrict__ cls1,
    const float* __restrict__ cls2, const float* __restrict__ dot_pg,
    const float* __restrict__ dot_sg, const float* __restrict__ sp_sum,
    const float* __restrict__ sig_sum, const float* __restrict__ gsum,
    const int* __restrict__ tgt, const float* __restrict__ matched,
    const float* __restrict__ msum, float* __restrict__ out) {
  const int tid = threadIdx.x;
  float acc = 0.f;
  for (int idx = tid; idx < NL * NB * NQ; idx += 256) {
    int l = idx / (NB * NQ);
    int rem = idx - l * NB * NQ;
    int b = rem / NQ;
    int q = rem - b * NQ;
    const float* cls = (l == 0) ? cls0 : (l == 1) ? cls1 : cls2;
    const float* lg = cls + (size_t)(b * NQ + q) * (NUM_CLASSES + 1);
    int   t = tgt[b * NQ + q];
    float m = matched[b * NQ + q];
    float mx = lg[0];
    for (int c = 1; c < NUM_CLASSES + 1; ++c) mx = fmaxf(mx, lg[c]);
    float se = 0.f;
    for (int c = 0; c < NUM_CLASSES + 1; ++c) se += fexp2((lg[c] - mx) * LOG2E);
    float ce = (mx + LN2 * flog2(se)) - lg[t];
    float n_unm = fmaxf((float)NQ - msum[b], 1.0f);
    float wl = (l == 2) ? 1.0f : 0.25f;  // final + AUX_W * mean(aux0, aux1)
    float contrib = 2.0f * ce * m + 0.1f * ce * (1.f - m) / n_unm;
    if (m > 0.5f && t < NUM_CLASSES) {
      size_t sb = (size_t)(l * NB + b) * QP + q;
      float bce  = (sp_sum[sb] - dot_pg[sb * CP + t]) * (1.0f / HWX);
      float dice = 1.f - (2.f * dot_sg[sb * CP + t] + 1e-5f) /
                         (sig_sum[sb] + gsum[b * CP + t] + 1e-5f);
      contrib += 5.f * bce + 5.f * dice;
    }
    acc += wl * contrib * (1.0f / NB);
  }
  __shared__ float red[256];
  red[tid] = acc;
  __syncthreads();
  for (int s = 128; s > 0; s >>= 1) {
    if (tid < s) red[tid] += red[tid + s];
    __syncthreads();
  }
  if (tid == 0) out[0] = red[0];
}

// ---------------------------------------------------------------------------
extern "C" void kernel_launch(void* const* d_in, const int* in_sizes, int n_in,
                              void* d_out, int out_size, void* d_ws,
                              size_t ws_size, hipStream_t stream) {
  (void)in_sizes; (void)n_in; (void)out_size; (void)ws_size;
  // setup_inputs() dict order: cls0, masks0, cls1, masks1, cls2, masks2, anno
  const float* cls0   = (const float*)d_in[0];
  const float* masks0 = (const float*)d_in[1];
  const float* cls1   = (const float*)d_in[2];
  const float* masks1 = (const float*)d_in[3];
  const float* cls2   = (const float*)d_in[4];
  const float* masks2 = (const float*)d_in[5];
  const int*   anno   = (const int*)d_in[6];
  float* out = (float*)d_out;

  // workspace layout (floats)
  float* ws      = (float*)d_ws;
  float* dot_pg  = ws;                         // NL*NB*QP*CP = 86016
  float* dot_sg  = dot_pg + (size_t)NL * NB * QP * CP;
  float* sp_sum  = dot_sg + (size_t)NL * NB * QP * CP;   // NL*NB*QP = 2688
  float* sig_sum = sp_sum + (size_t)NL * NB * QP;
  float* gsum    = sig_sum + (size_t)NL * NB * QP;       // NB*CP = 256
  float* matched = gsum + (size_t)NB * CP;               // NB*NQ = 800
  float* msum    = matched + (size_t)NB * NQ;            // NB
  int*   tgt     = (int*)(msum + NB);                    // NB*NQ ints

  gsum_kernel<<<NB, 256, 0, stream>>>(anno, gsum);
  stats_kernel<<<dim3(7, NB, NL), 256, 0, stream>>>(
      masks0, masks1, masks2, anno, dot_pg, dot_sg, sp_sum, sig_sum);
  match_kernel<<<NB, 128, 0, stream>>>(cls2, dot_pg, dot_sg, sp_sum, sig_sum,
                                       gsum, tgt, matched, msum);
  loss_kernel<<<1, 256, 0, stream>>>(cls0, cls1, cls2, dot_pg, dot_sg, sp_sum,
                                     sig_sum, gsum, tgt, matched, msum, out);
}
